// LatentSDE_41412074668601
// MI455X (gfx1250) — matrix operand order repaired and checked by reading, not compiled, not run on hardware
//
#include <hip/hip_runtime.h>
#include <hip/hip_bf16.h>
#include <math.h>

// ---------------------------------------------------------------------------
// Latent SDE forward for MI455X (gfx1250), wave32 + WMMA f32_16x16x32_f16.
// Weights held resident in VGPRs across the sequential time loops;
// global_prefetch_b8 used to hide next-step activation latency.
// ---------------------------------------------------------------------------

typedef _Float16 half_t;
typedef __attribute__((ext_vector_type(16))) _Float16 v16h;
typedef __attribute__((ext_vector_type(8)))  _Float16 v8h;
typedef __attribute__((ext_vector_type(8)))  float    v8f;

#define T_INN 100
#define T_OUTT 132
#define BATCH 1024
#define DD 32
#define LL 16
#define CC 64
#define HH 128
#define OO 32

// ---------------- WMMA fragment loaders (CDNA5 16x16x32 f16 layouts) --------
// A (16x32 f16): lanes 0-15 row M=lane, K = {0..7,16..23}; lanes 16-31 row
// M=lane-16, K = {8..15,24..31}  (ISA 7.12.2 table).
__device__ __forceinline__ v16h load_a_f16(const half_t* A, int lda, int r0, int k0) {
  const int lane = threadIdx.x & 31;
  const int m = lane & 15, hi = lane >> 4;
  const half_t* p = A + (size_t)(r0 + m) * lda + k0 + 8 * hi;
  v8h c0 = *(const v8h*)p;
  v8h c1 = *(const v8h*)(p + 16);
  v16h a;
#pragma unroll
  for (int j = 0; j < 8; ++j) { a[j] = c0[j]; a[8 + j] = c1[j]; }
  return a;
}

__device__ __forceinline__ v16h load_a_f32(const float* A, int lda, int r0, int k0) {
  const int lane = threadIdx.x & 31;
  const int m = lane & 15, hi = lane >> 4;
  const float* p = A + (size_t)(r0 + m) * lda + k0 + 8 * hi;
  v16h a;
#pragma unroll
  for (int j = 0; j < 8; ++j) { a[j] = (half_t)p[j]; a[8 + j] = (half_t)p[16 + j]; }
  return a;
}

// B (32x16 f16) from weight W (N x K row-major, so B = W^T): lane&15 = column
// N, lanes 0-15 K = {0..7,16..23}, lanes 16-31 K = {8..15,24..31} (matches A).
__device__ __forceinline__ v16h load_b_w(const half_t* W, int ldk, int n0, int k0) {
  const int lane = threadIdx.x & 31;
  const int n = lane & 15;
  const half_t* p = W + (size_t)(n0 + n) * ldk + k0 + ((lane >> 4) << 3);
  v8h c0 = *(const v8h*)p;
  v8h c1 = *(const v8h*)(p + 16);
  v16h b;
#pragma unroll
  for (int j = 0; j < 8; ++j) { b[j] = c0[j]; b[8 + j] = c1[j]; }
  return b;
}

__device__ __forceinline__ v8f wmma32(v16h a, v16h b, v8f c) {
  return __builtin_amdgcn_wmma_f32_16x16x32_f16(false, a, false, b, (short)0, c,
                                                false, false);
}

__device__ __forceinline__ float sigmoidf_(float x) { return 1.0f / (1.0f + expf(-x)); }
__device__ __forceinline__ float softplusf_(float x) { return x > 20.f ? x : log1pf(expf(x)); }
__device__ __forceinline__ float wave_sum(float v) {
#pragma unroll
  for (int o = 16; o > 0; o >>= 1) v += __shfl_down(v, o, 32);
  return v;
}

// ---------------- weight f32 -> f16 with K padding ---------------------------
__global__ __launch_bounds__(256) void cvt_pad_kernel(const float* __restrict__ src,
                                                      half_t* __restrict__ dst,
                                                      int N, int Kin, int Kpad) {
  int total = N * Kpad;
  for (int i = blockIdx.x * 256 + threadIdx.x; i < total; i += gridDim.x * 256) {
    int n = i / Kpad, k = i - n * Kpad;
    dst[i] = (k < Kin) ? (half_t)src[n * Kin + k] : (half_t)0.f;
  }
}

__global__ void init_kernel(float* accums) {
  if (threadIdx.x < 4) accums[threadIdx.x] = 0.f;
}

// ---------------- GRU: one direction of one layer ----------------------------
// Each WG owns 32 batch rows and loops all T steps (rows are independent).
// gates = [r | z | n], r/z fused over K=[x|h]; n split into i_n (x) + h_n (h).
// Wave j owns output h columns 16j..16j+15 (N-tiles j, j+8, j+16).
// W_hh fragments (12 x v16h = 96 VGPRs) stay resident across all 100 steps.
template <int IN, bool SRCF16, bool REV>
__global__ __launch_bounds__(256) void gru_kernel(
    const float* __restrict__ xs, const half_t* __restrict__ srcH,
    const half_t* __restrict__ wih, const half_t* __restrict__ whh,
    const float* __restrict__ bih, const float* __restrict__ bhh,
    half_t* __restrict__ yout, int ycol, float* __restrict__ hfin) {
  constexpr int KTOT = IN + HH;
  __shared__ half_t Abuf[32][KTOT];   // [x_t | h] f16, A operand
  __shared__ float  hbuf[32][HH];     // f32 hidden state (recurrence precision)
  const int tid = threadIdx.x, lane = tid & 31, j = tid >> 5;
  const int row0 = blockIdx.x * 32;
  const int col = (j << 4) + (lane & 15);
  for (int e = tid; e < 32 * HH; e += 256) {
    int r = e >> 7, c = e & 127;
    hbuf[r][c] = 0.f;
    Abuf[r][IN + c] = (half_t)0.f;
  }
  const float b_r  = bih[col] + bhh[col];
  const float b_z  = bih[HH + col] + bhh[HH + col];
  const float b_in = bih[2 * HH + col];
  const float b_hn = bhh[2 * HH + col];
  // resident W_hh fragments (loop-invariant across all time steps)
  v16h whf[HH / 32][3];
#pragma unroll
  for (int kt = 0; kt < HH / 32; ++kt) {
    whf[kt][0] = load_b_w(whh, HH, (j << 4),          kt * 32);
    whf[kt][1] = load_b_w(whh, HH, HH + (j << 4),     kt * 32);
    whf[kt][2] = load_b_w(whh, HH, 2 * HH + (j << 4), kt * 32);
  }
  // for the K=32 input layer, W_ih fits in registers too (3 fragments)
  v16h wif[3] = {};
  if (IN == 32) {
    wif[0] = load_b_w(wih, IN, (j << 4), 0);
    wif[1] = load_b_w(wih, IN, HH + (j << 4), 0);
    wif[2] = load_b_w(wih, IN, 2 * HH + (j << 4), 0);
  }
  for (int s = 0; s < T_INN; ++s) {
    const int t = REV ? (T_INN - 1 - s) : s;
    if (SRCF16) {
      for (int e = tid; e < 32 * IN; e += 256) {
        int r = e / IN, c = e - r * IN;
        Abuf[r][c] = srcH[((size_t)t * BATCH + row0 + r) * IN + c];
      }
    } else {
      // layer-0 input is x_rev[t] = xs[T-1-t]
      for (int e = tid; e < 32 * IN; e += 256) {
        int r = e / IN, c = e - r * IN;
        Abuf[r][c] = (half_t)xs[((size_t)(T_INN - 1 - t) * BATCH + row0 + r) * DD + c];
      }
    }
    // prefetch next step's input slab (global_prefetch_b8)
    if (s + 1 < T_INN) {
      const int tn = REV ? (T_INN - 2 - s) : (s + 1);
      if (SRCF16) {
        const char* pf = (const char*)(srcH + ((size_t)tn * BATCH + row0) * IN);
        if (tid * 64 < 32 * IN * 2) __builtin_prefetch(pf + tid * 64, 0, 1);
      } else {
        const char* pf = (const char*)(xs + ((size_t)(T_INN - 1 - tn) * BATCH + row0) * DD);
        if (tid * 16 < 32 * DD * 4) __builtin_prefetch(pf + tid * 16, 0, 1);
      }
    }
    __syncthreads();
    float hnew[2][8];
#pragma unroll
    for (int mt = 0; mt < 2; ++mt) {
      v8f ar = {}, az = {}, ain = {}, ahn = {};
#pragma unroll
      for (int kt = 0; kt < IN / 32; ++kt) {
        v16h a = load_a_f16(&Abuf[0][0], KTOT, mt * 16, kt * 32);
        if (IN == 32) {
          ar  = wmma32(a, wif[0], ar);
          az  = wmma32(a, wif[1], az);
          ain = wmma32(a, wif[2], ain);
        } else {
          ar  = wmma32(a, load_b_w(wih, IN, (j << 4),          kt * 32), ar);
          az  = wmma32(a, load_b_w(wih, IN, HH + (j << 4),     kt * 32), az);
          ain = wmma32(a, load_b_w(wih, IN, 2 * HH + (j << 4), kt * 32), ain);
        }
      }
#pragma unroll
      for (int kt = 0; kt < HH / 32; ++kt) {
        v16h a = load_a_f16(&Abuf[0][0], KTOT, mt * 16, IN + kt * 32);
        ar  = wmma32(a, whf[kt][0], ar);
        az  = wmma32(a, whf[kt][1], az);
        ahn = wmma32(a, whf[kt][2], ahn);
      }
      const int rb = mt * 16 + ((lane >> 4) << 3);
#pragma unroll
      for (int vr = 0; vr < 8; ++vr) {
        float hp = hbuf[rb + vr][col];
        float rg = sigmoidf_(ar[vr] + b_r);
        float zg = sigmoidf_(az[vr] + b_z);
        float ng = tanhf(ain[vr] + b_in + rg * (ahn[vr] + b_hn));
        hnew[mt][vr] = (1.f - zg) * ng + zg * hp;
      }
    }
    __syncthreads();  // all A/h reads done before h update
#pragma unroll
    for (int mt = 0; mt < 2; ++mt) {
      const int rb = mt * 16 + ((lane >> 4) << 3);
#pragma unroll
      for (int vr = 0; vr < 8; ++vr) {
        int r = rb + vr;
        float v = hnew[mt][vr];
        hbuf[r][col] = v;
        Abuf[r][IN + col] = (half_t)v;
        yout[((size_t)t * BATCH + row0 + r) * (2 * HH) + ycol + col] = (half_t)v;
      }
    }
  }
  __syncthreads();
  for (int e = tid; e < 32 * HH; e += 256) {
    int r = e >> 7, c = e & 127;
    hfin[(size_t)(row0 + r) * HH + c] = hbuf[r][c];
  }
}

// ---------------- qz0: fin @ qz0_w^T, z0 = mean + exp(ls)*eps, KL ------------
__global__ __launch_bounds__(256) void qz0_kernel(
    const float* __restrict__ hfin, const half_t* __restrict__ qw,
    const float* __restrict__ qb, const float* __restrict__ eps0,
    const float* __restrict__ pz0m, const float* __restrict__ pz0ls,
    float* __restrict__ zs, float* __restrict__ accums) {
  __shared__ float q[64][32];
  const int tid = threadIdx.x, lane = tid & 31, j = tid >> 5;
  const int row0 = blockIdx.x * 64;
  const int mt = j & 3, nt = j >> 2;
  v8f acc = {};
#pragma unroll
  for (int kt = 0; kt < 16; ++kt) {
    int k0 = kt * 32;
    const float* base = hfin + (size_t)(k0 >> 7) * BATCH * HH;  // fin blocks
    acc = wmma32(load_a_f32(base + (size_t)row0 * HH, HH, mt * 16, k0 & 127),
                 load_b_w(qw, 512, nt * 16, k0), acc);
  }
  const int c = nt * 16 + (lane & 15);
  const float bias = qb[c];
  const int rb = mt * 16 + ((lane >> 4) << 3);
#pragma unroll
  for (int vr = 0; vr < 8; ++vr) q[rb + vr][c] = acc[vr] + bias;
  __syncthreads();
  float kls = 0.f;
  for (int e = tid; e < 64 * 16; e += 256) {
    int r = e >> 4, cl = e & 15;
    float mean = q[r][cl], ls = q[r][cl + 16];
    float z0 = mean + expf(ls) * eps0[(size_t)(row0 + r) * LL + cl];
    zs[(size_t)(row0 + r) * LL + cl] = z0;
    float pm = pz0m[cl], pls = pz0ls[cl], d = mean - pm;
    kls += (pls - ls) + 0.5f * (expf(2.f * (ls - pls)) + d * d * expf(-2.f * pls) - 1.f);
  }
  kls = wave_sum(kls);
  if (lane == 0) atomicAdd(&accums[1], kls);
}

// ---------------- ctx = flip(seq @ lin_w^T + lin_b, time) --------------------
__global__ __launch_bounds__(256) void ctx_kernel(const half_t* __restrict__ seq,
                                                  const half_t* __restrict__ lw,
                                                  const float* __restrict__ lb,
                                                  half_t* __restrict__ ctxh) {
  const int t = blockIdx.x >> 4;
  const int chunk = blockIdx.x & 15;
  const int lane = threadIdx.x & 31, j = threadIdx.x >> 5;
  const int nt = j & 3, mtb = (j >> 2) << 1;
  const half_t* Arow = seq + ((size_t)t * BATCH + chunk * 64) * 256;
  const int c = nt * 16 + (lane & 15);
  const float bias = lb[c];
#pragma unroll
  for (int mi = 0; mi < 2; ++mi) {
    int mt = mtb + mi;
    v8f acc = {};
#pragma unroll
    for (int kt = 0; kt < 8; ++kt)
      acc = wmma32(load_a_f16(Arow, 256, mt * 16, kt * 32),
                   load_b_w(lw, 256, nt * 16, kt * 32), acc);
    const int rb = mt * 16 + ((lane >> 4) << 3);
#pragma unroll
    for (int vr = 0; vr < 8; ++vr) {
      int r = chunk * 64 + rb + vr;
      ctxh[((size_t)(T_INN - 1 - t) * BATCH + r) * CC + c] = (half_t)(acc[vr] + bias);
    }
  }
}

// ---------------- SDE MLP helpers -------------------------------------------
// 64 rows x 128 cols, softplus, N-tile = wave id; B fragments pre-loaded.
template <int KT, int LDA>
__device__ __forceinline__ void mlp_layer_sp_f(const half_t* A, const v16h* bf,
                                               float bc, half_t (*out)[128]) {
  const int lane = threadIdx.x & 31, j = threadIdx.x >> 5;
  const int c = (j << 4) + (lane & 15);
#pragma unroll
  for (int mt = 0; mt < 4; ++mt) {
    v8f acc = {};
#pragma unroll
    for (int kt = 0; kt < KT; ++kt)
      acc = wmma32(load_a_f16(A, LDA, mt * 16, kt * 32), bf[kt], acc);
    const int rb = mt * 16 + ((lane >> 4) << 3);
#pragma unroll
    for (int vr = 0; vr < 8; ++vr) out[rb + vr][c] = (half_t)softplusf_(acc[vr] + bc);
  }
}

// 64 rows x 16 cols final layer (waves 0..3 only), K=128, streamed weights.
__device__ __forceinline__ void mlp_layer_out(const half_t* A, const half_t* W,
                                              const float* bias, float (*out)[16],
                                              bool sigm) {
  const int lane = threadIdx.x & 31, j = threadIdx.x >> 5;
  if (j < 4) {
    const int c = lane & 15;
    const float b = bias[c];
    v8f acc = {};
#pragma unroll
    for (int kt = 0; kt < 4; ++kt)
      acc = wmma32(load_a_f16(A, 128, j * 16, kt * 32),
                   load_b_w(W, 128, 0, kt * 32), acc);
    const int rb = j * 16 + ((lane >> 4) << 3);
#pragma unroll
    for (int vr = 0; vr < 8; ++vr) {
      float v = acc[vr] + b;
      out[rb + vr][c] = sigm ? sigmoidf_(v) : v;
    }
  }
}

// ---------------- Euler-Maruyama SDE loop (persistent per 64-row WG) --------
__global__ __launch_bounds__(256) void sde_kernel(
    const float* __restrict__ tsin, const float* __restrict__ tsout,
    const float* __restrict__ dW, const half_t* __restrict__ ctxh,
    const half_t* fw0, const float* fb0, const half_t* fw1, const float* fb1,
    const half_t* fw2, const float* fb2,
    const half_t* hw0, const float* hb0, const half_t* hw1, const float* hb1,
    const half_t* hw2, const float* hb2,
    const half_t* gw0, const float* gb0, const half_t* gw1, const float* gb1,
    float* __restrict__ zs, float* __restrict__ accums) {
  __shared__ half_t zc[64][96];     // [z | ctx | pad]  f-net input (K=80->96)
  __shared__ half_t zh[64][32];     // [z | pad]        h/g-net input (K=16->32)
  __shared__ half_t act[64][128];
  __shared__ half_t act2[64][128];
  __shared__ float fde[64][16];
  __shared__ float hde[64][16];
  __shared__ float gde[64][16];
  const int tid = threadIdx.x, lane = tid & 31, j = tid >> 5;
  const int row0 = blockIdx.x * 64;
  const int c128 = (j << 4) + (lane & 15);
  float zloc[4];
#pragma unroll
  for (int qd = 0; qd < 4; ++qd) {
    int e = (tid << 2) + qd;
    zloc[qd] = zs[(size_t)(row0 + (e >> 4)) * LL + (e & 15)];
  }
  for (int e = tid; e < 64 * 16; e += 256) zc[e >> 4][80 + (e & 15)] = (half_t)0.f;
  for (int e = tid; e < 64 * 16; e += 256) zh[e >> 4][16 + (e & 15)] = (half_t)0.f;
  // resident softplus-layer weight fragments (loop-invariant, ~104 VGPRs)
  v16h f0f[3], f1f[4], h1f[4], h0f1[1], g0f1[1];
#pragma unroll
  for (int kt = 0; kt < 3; ++kt) f0f[kt] = load_b_w(fw0, 96, (j << 4), kt * 32);
#pragma unroll
  for (int kt = 0; kt < 4; ++kt) f1f[kt] = load_b_w(fw1, 128, (j << 4), kt * 32);
#pragma unroll
  for (int kt = 0; kt < 4; ++kt) h1f[kt] = load_b_w(hw1, 128, (j << 4), kt * 32);
  h0f1[0] = load_b_w(hw0, 32, (j << 4), 0);
  g0f1[0] = load_b_w(gw0, 32, (j << 4), 0);
  const float fb0c = fb0[c128], fb1c = fb1[c128];
  const float hb0c = hb0[c128], hb1c = hb1[c128], gb0c = gb0[c128];
  float lq = 0.f;
  for (int s = 0; s < T_OUTT - 1; ++s) {
    const float t = tsout[s];
    const float dt = tsout[s + 1] - t;
    int lo = 0, hi = T_INN;                // searchsorted(ts_in, t, 'right')
    while (lo < hi) { int mid = (lo + hi) >> 1; if (tsin[mid] <= t) lo = mid + 1; else hi = mid; }
    const int ci = lo < (T_INN - 1) ? lo : (T_INN - 1);
#pragma unroll
    for (int qd = 0; qd < 4; ++qd) {       // stage z (owner-thread writes)
      int e = (tid << 2) + qd; int r = e >> 4, c = e & 15;
      half_t hz = (half_t)zloc[qd];
      zc[r][c] = hz; zh[r][c] = hz;
    }
    for (int e = tid; e < 64 * CC; e += 256) {   // stage ctx[ci]
      int r = e >> 6, c = e & 63;
      zc[r][16 + c] = ctxh[((size_t)ci * BATCH + row0 + r) * CC + c];
    }
    // prefetch next step's dW block and ctx tile (global_prefetch_b8)
    if (s + 1 < T_OUTT - 1) {
      __builtin_prefetch((const char*)(dW + ((size_t)(s + 1) * BATCH + row0) * LL) + tid * 16, 0, 1);
      const int cin = (s + 2 < T_INN) ? (s + 2) : (T_INN - 1);
      __builtin_prefetch((const char*)(ctxh + ((size_t)cin * BATCH + row0) * CC) + tid * 32, 0, 1);
    }
    __syncthreads();
    mlp_layer_sp_f<3, 96>(&zc[0][0], f0f, fb0c, act);     __syncthreads();
    mlp_layer_sp_f<4, 128>(&act[0][0], f1f, fb1c, act2);  __syncthreads();
    mlp_layer_out(&act2[0][0], fw2, fb2, fde, false);     // f output
    mlp_layer_sp_f<1, 32>(&zh[0][0], h0f1, hb0c, act);    __syncthreads();
    mlp_layer_sp_f<4, 128>(&act[0][0], h1f, hb1c, act2);  __syncthreads();
    mlp_layer_out(&act2[0][0], hw2, hb2, hde, false);     // h output
    mlp_layer_sp_f<1, 32>(&zh[0][0], g0f1, gb0c, act);    __syncthreads();
    mlp_layer_out(&act[0][0], gw1, gb1, gde, true);       // g output (sigmoid)
    __syncthreads();
#pragma unroll
    for (int qd = 0; qd < 4; ++qd) {
      int e = (tid << 2) + qd; int r = e >> 4, c = e & 15;
      float fv = fde[r][c], hv = hde[r][c], gv = gde[r][c];
      float u = (fv - hv) / gv;
      lq += 0.5f * u * u * dt;
      float dw = dW[((size_t)s * BATCH + row0 + r) * LL + c];
      zloc[qd] = zloc[qd] + fv * dt + gv * dw;
      zs[((size_t)(s + 1) * BATCH + row0 + r) * LL + c] = zloc[qd];
    }
    __syncthreads();
  }
  lq = wave_sum(lq);
  if (lane == 0) atomicAdd(&accums[2], lq);
}

// ---------------- decoder likelihood: mu/sigma heads + reduce ----------------
__global__ __launch_bounds__(256) void lik_kernel(
    const float* __restrict__ zs, const float* __restrict__ xs,
    const half_t* __restrict__ muw, const float* __restrict__ mub,
    const half_t* __restrict__ lsw, const float* __restrict__ lsb,
    float* __restrict__ accums) {
  __shared__ half_t zpad[64][32];
  __shared__ float mu[64][32];
  __shared__ float sg[64][32];
  const int tid = threadIdx.x, lane = tid & 31, j = tid >> 5;
  const size_t grow0 = (size_t)blockIdx.x * 64;
  for (int e = tid; e < 64 * 32; e += 256) {
    int r = e >> 5, c = e & 31;
    zpad[r][c] = (c < 16) ? (half_t)zs[(grow0 + r) * LL + c] : (half_t)0.f;
  }
  __syncthreads();
#pragma unroll
  for (int tt = 0; tt < 2; ++tt) {
    int idx = j + tt * 8;
    int mat = idx >> 3, rem = idx & 7, mt = rem & 3, nt = rem >> 2;
    const half_t* W = mat ? lsw : muw;
    const float* bb = mat ? lsb : mub;
    v8f acc = {};
    acc = wmma32(load_a_f16(&zpad[0][0], 32, mt * 16, 0), load_b_w(W, 32, nt * 16, 0), acc);
    const int c = nt * 16 + (lane & 15);
    const float b = bb[c];
    const int rb = mt * 16 + ((lane >> 4) << 3);
#pragma unroll
    for (int vr = 0; vr < 8; ++vr) {
      float v = acc[vr] + b;
      if (mat) sg[rb + vr][c] = expf(fminf(fmaxf(v, -10.f), 2.f));
      else mu[rb + vr][c] = v;
    }
  }
  __syncthreads();
  float lp = 0.f;
  for (int e = tid; e < 64 * 32; e += 256) {
    int r = e >> 5, c = e & 31;
    float x = xs[(grow0 + r) * DD + c];
    float d = (x - mu[r][c]) / sg[r][c];
    lp += -0.5f * d * d - logf(sg[r][c]) - 0.91893853320467274f;  // 0.5*log(2*pi)
  }
  lp = wave_sum(lp);
  if (lane == 0) atomicAdd(&accums[0], lp);
}

// ---------------- output head: zs[100:] @ out_w^T + out_b --------------------
__global__ __launch_bounds__(256) void out_kernel(const float* __restrict__ zst,
                                                  const half_t* __restrict__ ow,
                                                  const float* __restrict__ ob,
                                                  float* __restrict__ out) {
  __shared__ half_t zpad[64][32];
  const int tid = threadIdx.x, lane = tid & 31, j = tid >> 5;
  const size_t grow0 = (size_t)blockIdx.x * 64;
  for (int e = tid; e < 64 * 32; e += 256) {
    int r = e >> 5, c = e & 31;
    zpad[r][c] = (c < 16) ? (half_t)zst[(grow0 + r) * LL + c] : (half_t)0.f;
  }
  __syncthreads();
  const int mt = j & 3, nt = j >> 2;
  v8f acc = {};
  acc = wmma32(load_a_f16(&zpad[0][0], 32, mt * 16, 0), load_b_w(ow, 32, nt * 16, 0), acc);
  const int c = nt * 16 + (lane & 15);
  const float b = ob[c];
  const int rb = mt * 16 + ((lane >> 4) << 3);
#pragma unroll
  for (int vr = 0; vr < 8; ++vr)
    out[(grow0 + rb + vr) * OO + c] = acc[vr] + b;
}

__global__ void finalize_kernel(const float* accums, float* out_tail) {
  out_tail[0] = accums[0] / (float)BATCH;                              // log_pxs
  out_tail[1] = accums[1] / (float)BATCH + accums[2] / (float)BATCH;   // logqp0 + path
}

// ---------------------------------------------------------------------------
extern "C" void kernel_launch(void* const* d_in, const int* in_sizes, int n_in,
                              void* d_out, int out_size, void* d_ws, size_t ws_size,
                              hipStream_t stream) {
  (void)in_sizes; (void)out_size;
  const float* xs    = (const float*)d_in[0];
  const float* tsin  = (const float*)d_in[1];
  const float* tsout = (const float*)d_in[2];
  const float* eps0  = (const float*)d_in[3];
  const float* dW    = (const float*)d_in[4];
  const float* P[44];
  for (int i = 0; i < 44 && (5 + i) < n_in; ++i) P[i] = (const float*)d_in[5 + i];
  // P: 0-3 gru0f(wih,bih,whh,bhh) 4-7 gru0b 8-11 gru1f 12-15 gru1b
  //    16 qz0_w 17 qz0_b 18 lin_w 19 lin_b 20-25 f(w0,b0,w1,b1,w2,b2)
  //    26-31 h  32-35 g(w0,b0,w1,b1) 36 mu_w 37 mu_b 38 ls_w 39 ls_b
  //    40 out_w 41 out_b 42 pz0_mean 43 pz0_logstd

  char* w = (char*)d_ws;
  size_t off = 0;
  auto alloc_h = [&](size_t n) { half_t* p = (half_t*)(w + off); off += ((n * 2 + 255) & ~(size_t)255); return p; };
  auto alloc_f = [&](size_t n) { float*  p = (float*)(w + off); off += ((n * 4 + 255) & ~(size_t)255); return p; };

  half_t* w0f_ih = alloc_h(384 * 32);   half_t* w0f_hh = alloc_h(384 * 128);
  half_t* w0b_ih = alloc_h(384 * 32);   half_t* w0b_hh = alloc_h(384 * 128);
  half_t* w1f_ih = alloc_h(384 * 256);  half_t* w1f_hh = alloc_h(384 * 128);
  half_t* w1b_ih = alloc_h(384 * 256);  half_t* w1b_hh = alloc_h(384 * 128);
  half_t* qz0w = alloc_h(32 * 512);     half_t* linw = alloc_h(64 * 256);
  half_t* fw0 = alloc_h(128 * 96);      half_t* fw1 = alloc_h(128 * 128);
  half_t* fw2 = alloc_h(16 * 128);
  half_t* hw0 = alloc_h(128 * 32);      half_t* hw1 = alloc_h(128 * 128);
  half_t* hw2 = alloc_h(16 * 128);
  half_t* gw0 = alloc_h(128 * 32);      half_t* gw1 = alloc_h(16 * 128);
  half_t* muw = alloc_h(32 * 32);       half_t* lsw = alloc_h(32 * 32);
  half_t* outw = alloc_h(32 * 32);
  half_t* l0h  = alloc_h((size_t)T_INN * BATCH * 256);
  half_t* seqh = alloc_h((size_t)T_INN * BATCH * 256);
  half_t* ctxh = alloc_h((size_t)T_INN * BATCH * CC);
  float*  hfin = alloc_f((size_t)4 * BATCH * HH);
  float*  zs   = alloc_f((size_t)T_OUTT * BATCH * LL);
  float*  acc  = alloc_f(4);
  if (off > ws_size) return;  // workspace too small: avoid OOB

  init_kernel<<<1, 64, 0, stream>>>(acc);

  auto cvt = [&](const float* s, half_t* d, int N, int Kin, int Kpad) {
    int total = N * Kpad;
    cvt_pad_kernel<<<(total + 255) / 256, 256, 0, stream>>>(s, d, N, Kin, Kpad);
  };
  cvt(P[0],  w0f_ih, 384, 32, 32);   cvt(P[2],  w0f_hh, 384, 128, 128);
  cvt(P[4],  w0b_ih, 384, 32, 32);   cvt(P[6],  w0b_hh, 384, 128, 128);
  cvt(P[8],  w1f_ih, 384, 256, 256); cvt(P[10], w1f_hh, 384, 128, 128);
  cvt(P[12], w1b_ih, 384, 256, 256); cvt(P[14], w1b_hh, 384, 128, 128);
  cvt(P[16], qz0w, 32, 512, 512);    cvt(P[18], linw, 64, 256, 256);
  cvt(P[20], fw0, 128, 80, 96);      cvt(P[22], fw1, 128, 128, 128);
  cvt(P[24], fw2, 16, 128, 128);
  cvt(P[26], hw0, 128, 16, 32);      cvt(P[28], hw1, 128, 128, 128);
  cvt(P[30], hw2, 16, 128, 128);
  cvt(P[32], gw0, 128, 16, 32);      cvt(P[34], gw1, 16, 128, 128);
  cvt(P[36], muw, 32, 16, 32);       cvt(P[38], lsw, 32, 16, 32);
  cvt(P[40], outw, 32, 16, 32);

  // encoder (sequence reversed inside layer-0 kernels)
  gru_kernel<32, false, false><<<32, 256, 0, stream>>>(xs, nullptr, w0f_ih, w0f_hh,
      P[1], P[3], l0h, 0, hfin + 0 * (size_t)BATCH * HH);
  gru_kernel<32, false, true><<<32, 256, 0, stream>>>(xs, nullptr, w0b_ih, w0b_hh,
      P[5], P[7], l0h, 128, hfin + 1 * (size_t)BATCH * HH);
  gru_kernel<256, true, false><<<32, 256, 0, stream>>>(nullptr, l0h, w1f_ih, w1f_hh,
      P[9], P[11], seqh, 0, hfin + 2 * (size_t)BATCH * HH);
  gru_kernel<256, true, true><<<32, 256, 0, stream>>>(nullptr, l0h, w1b_ih, w1b_hh,
      P[13], P[15], seqh, 128, hfin + 3 * (size_t)BATCH * HH);

  qz0_kernel<<<16, 256, 0, stream>>>(hfin, qz0w, P[17], eps0, P[42], P[43], zs, acc);
  ctx_kernel<<<T_INN * 16, 256, 0, stream>>>(seqh, linw, P[19], ctxh);
  sde_kernel<<<16, 256, 0, stream>>>(tsin, tsout, dW, ctxh,
      fw0, P[21], fw1, P[23], fw2, P[25],
      hw0, P[27], hw1, P[29], hw2, P[31],
      gw0, P[33], gw1, P[35], zs, acc);
  lik_kernel<<<T_INN * 16, 256, 0, stream>>>(zs, xs, muw, P[37], lsw, P[39], acc);
  out_kernel<<<(T_OUTT - T_INN) * 16, 256, 0, stream>>>(
      zs + (size_t)T_INN * BATCH * LL, outw, P[41], (float*)d_out);
  finalize_kernel<<<1, 1, 0, stream>>>(acc, (float*)d_out + (size_t)(T_OUTT - T_INN) * BATCH * OO);
}
